// RecurrentTokenizer_69664369541337
// MI455X (gfx1250) — matrix operand order repaired and checked by reading, not compiled, not run on hardware
//
#include <hip/hip_runtime.h>
#include <hip/hip_bf16.h>

typedef __attribute__((ext_vector_type(16))) __bf16 v16bf;
typedef __attribute__((ext_vector_type(8)))  float  v8f;

#define N_   32
#define HW_  16384
#define C_   256
#define L_   16
#define D_   128
#define BP   256            // pixels per block
#define NBLK (HW_ / BP)     // 64 pixel-blocks per batch item

// ---------------- fragment helpers (CDNA5 wave32 WMMA layouts) ----------------
// 16-bit A-matrix 16x32: lanes 0-15 row M=r hold K = c0+0..7 (v0-3) and c0+16..23 (v4-7);
// lanes 16-31 hold K = c0+8..15 and c0+24..31.
__device__ __forceinline__ v16bf load_a_bf16(const __bf16* row, int c0, int hv) {
  v16bf f;
  const __bf16* p0 = row + c0 + hv * 8;
#pragma unroll
  for (int e = 0; e < 8; ++e) f[e] = p0[e];
#pragma unroll
  for (int e = 0; e < 8; ++e) f[8 + e] = p0[16 + e];
  return f;
}
// 16-bit B-matrix 32x16: lane r holds column N=r; lanes 0-15 K=c0+0..7 and c0+16..23, lanes 16-31 K=c0+8..15 and c0+24..31.
__device__ __forceinline__ v16bf load_b_bf16(const __bf16* col, int c0, int hv) {
  v16bf f;
  const __bf16* p = col + c0 + hv * 8;
#pragma unroll
  for (int e = 0; e < 8; ++e) f[e] = p[e];
#pragma unroll
  for (int e = 0; e < 8; ++e) f[8 + e] = p[16 + e];
  return f;
}
// Convert staged raw fp32 x data (two v8f: K lo-quad, K hi-quad) into an A fragment.
__device__ __forceinline__ v16bf cvt_frag(const v8f* s) {
  v16bf f;
#pragma unroll
  for (int e = 0; e < 8; ++e) f[e] = (__bf16)s[0][e];
#pragma unroll
  for (int e = 0; e < 8; ++e) f[8 + e] = (__bf16)s[1][e];
  return f;
}

#define WMMA_BF16(A, B, Cacc) \
  __builtin_amdgcn_wmma_f32_16x16x32_bf16(false, (A), false, (B), (short)0, (Cacc), false, false)

// ---------------- prep kernels ----------------
// a[n,l,e] = sum_d t[n,l,d] * W1[e,d] + b1[e]   -> bf16, row-major [n][l][e]
__global__ void prep_a_kernel(const float* __restrict__ t, const float* __restrict__ W1,
                              const float* __restrict__ b1, __bf16* __restrict__ a_bf) {
  int idx = blockIdx.x * blockDim.x + threadIdx.x;   // 0 .. 65535
  int e  = idx & (D_ - 1);
  int nl = idx >> 7;
  const float* trow = t + (size_t)nl * D_;
  const float* wrow = W1 + (size_t)e * D_;
  float s = b1[e];
#pragma unroll 8
  for (int d = 0; d < D_; ++d) s = fmaf(trow[d], wrow[d], s);
  a_bf[idx] = (__bf16)s;
}

__global__ void conv_w2_kernel(const float* __restrict__ W2, __bf16* __restrict__ w2bf) {
  int i = blockIdx.x * blockDim.x + threadIdx.x;     // 0 .. 32767
  w2bf[i] = (__bf16)W2[i];
}

// ---------------- main fused kernel ----------------
__global__ void __launch_bounds__(256, 1)
main_kernel(const float* __restrict__ x, const float* __restrict__ b2v,
            const __bf16* __restrict__ w2bf, const __bf16* __restrict__ a_bf,
            float* __restrict__ pm, float* __restrict__ pz, float* __restrict__ pacc) {
  // LDS (strides padded so 16-lane row accesses step 4 banks -> conflict-free)
  __shared__ __bf16 w2s[D_][C_ + 8];      // W2 staged once per block (B-role, GEMM1)
  __shared__ __bf16 b_pd[BP][D_ + 8];     // b, [pixel][d]      (A-role for scores)
  __shared__ __bf16 b_dp[D_][BP + 8];     // b, [d][pixel]      (B-role for output)
  __shared__ float  s_t[L_][BP + 1];      // scores [l][pixel]
  __shared__ __bf16 e_t[L_][BP + 8];      // exp(s-m) [l][pixel] (A-role for output)
  __shared__ float  red_m[16][L_];
  __shared__ float  red_z[16][L_];
  __shared__ float  m_l[L_];

  const int n    = blockIdx.y;
  const int blk  = blockIdx.x;
  const int tid  = threadIdx.x;
  const int wv   = tid >> 5;
  const int lane = tid & 31;
  const int r    = lane & 15;
  const int hv   = lane >> 4;

  // ---- Stage W2 (bf16) into LDS: each thread copies half a row (128 bf16 = 256 B) ----
  {
    const int d = tid >> 1;
    const int c = (tid & 1) * 128;
    const uint4* src = (const uint4*)(w2bf + (size_t)d * C_ + c);
    uint4* dst = (uint4*)&w2s[d][c];
#pragma unroll
    for (int i = 0; i < 16; ++i) dst[i] = src[i];
  }
  __syncthreads();

  // ---- Phase A: b = x @ W2^T : wave owns 32 pixels (2 M-tiles), K = C = 256 ----
  const int p0 = wv * 32;
  const float* xrow0 = x + ((size_t)n * HW_ + (size_t)blk * BP + p0 + r) * C_ + hv * 8;
  const float* xrow1 = xrow0 + 16 * C_;

  v8f acc[2][8];
#pragma unroll
  for (int mt = 0; mt < 2; ++mt)
#pragma unroll
    for (int nt = 0; nt < 8; ++nt) acc[mt][nt] = (v8f){};

  // raw x staging: [mt*2 + part], part0 = K+0..7, part1 = K+16..23 (per half-wave)
  v8f buf0[4], buf1[4];
#define LOAD_X(BUF, c0)                                  \
  do {                                                   \
    (BUF)[0] = *(const v8f*)(xrow0 + (c0));              \
    (BUF)[1] = *(const v8f*)(xrow0 + (c0) + 16);         \
    (BUF)[2] = *(const v8f*)(xrow1 + (c0));              \
    (BUF)[3] = *(const v8f*)(xrow1 + (c0) + 16);         \
  } while (0)

#define GEMM_STEP(BUF, c0)                                                        \
  do {                                                                            \
    v16bf a0 = cvt_frag(&(BUF)[0]);                                               \
    v16bf a1 = cvt_frag(&(BUF)[2]);                                               \
    _Pragma("unroll")                                                             \
    for (int h = 0; h < 2; ++h) {                                                 \
      v16bf bb[4];                                                                \
      _Pragma("unroll")                                                           \
      for (int q = 0; q < 4; ++q)                                                 \
        bb[q] = load_b_bf16(&w2s[(h * 4 + q) * 16 + r][0], (c0), hv);             \
      _Pragma("unroll")                                                           \
      for (int q = 0; q < 4; ++q) {                                               \
        acc[0][h * 4 + q] = WMMA_BF16(a0, bb[q], acc[0][h * 4 + q]);              \
        acc[1][h * 4 + q] = WMMA_BF16(a1, bb[q], acc[1][h * 4 + q]);              \
      }                                                                           \
    }                                                                             \
  } while (0)

  LOAD_X(buf0, 0);
#pragma unroll 1
  for (int kk = 0; kk < 4; ++kk) {
    const int k0 = kk * 64;
    LOAD_X(buf1, k0 + 32);                       // next chunk in flight
    GEMM_STEP(buf0, k0);
    const int k2 = (kk < 3) ? (k0 + 64) : 0;     // branchless wrap (dummy reload on last)
    LOAD_X(buf0, k2);
    GEMM_STEP(buf1, k0 + 32);
  }

  // ---- Phase B/C: add bias, stash b into LDS (both orientations) as bf16 ----
#pragma unroll
  for (int nt = 0; nt < 8; ++nt) {
    const int d = nt * 16 + r;
    const float bias = b2v[d];
#pragma unroll
    for (int mt = 0; mt < 2; ++mt)
#pragma unroll
      for (int j = 0; j < 8; ++j) {
        const int p = p0 + mt * 16 + j + hv * 8;   // C-layout: VGPR j -> row j / j+8
        const __bf16 vb = (__bf16)(acc[mt][nt][j] + bias);
        b_pd[p][d] = vb;
        b_dp[d][p] = vb;
      }
  }
  __syncthreads();

  // ---- Phase D: scores s[p][l] = b[p][:] . a[l][:]  (M=pixels, N=L=16, K=D=128) ----
  v8f sc[2];
  sc[0] = (v8f){};
  sc[1] = (v8f){};
#pragma unroll
  for (int k = 0; k < 4; ++k) {
    const int k0 = k * 32;
    v16bf aa0 = load_a_bf16(&b_pd[p0 + r][0], k0, hv);
    v16bf aa1 = load_a_bf16(&b_pd[p0 + 16 + r][0], k0, hv);
    v16bf bb  = load_b_bf16(a_bf + ((size_t)n * L_ + r) * D_, k0, hv);
    sc[0] = WMMA_BF16(aa0, bb, sc[0]);
    sc[1] = WMMA_BF16(aa1, bb, sc[1]);
  }
#pragma unroll
  for (int mt = 0; mt < 2; ++mt)
#pragma unroll
    for (int j = 0; j < 8; ++j) {
      const int p = p0 + mt * 16 + j + hv * 8;
      s_t[r][p] = sc[mt][j];   // [l = lane%16][pixel]
    }
  __syncthreads();

  // ---- Phase E: block-local softmax stats over the 256 pixels ----
  const int sl  = tid & 15;   // l
  const int seg = tid >> 4;   // 16 pixel-segments of 16
  float mloc = -3.4e38f;
#pragma unroll
  for (int i = 0; i < 16; ++i) mloc = fmaxf(mloc, s_t[sl][seg * 16 + i]);
  red_m[seg][sl] = mloc;
  __syncthreads();
  if (tid < L_) {
    float m = red_m[0][tid];
#pragma unroll
    for (int g = 1; g < 16; ++g) m = fmaxf(m, red_m[g][tid]);
    m_l[tid] = m;
  }
  __syncthreads();
  const float mm = m_l[sl];
  float zloc = 0.f;
#pragma unroll
  for (int i = 0; i < 16; ++i) {
    const int p = seg * 16 + i;
    const float ev = __expf(s_t[sl][p] - mm);
    zloc += ev;
    e_t[sl][p] = (__bf16)ev;
  }
  red_z[seg][sl] = zloc;
  __syncthreads();
  if (tid < L_) {
    float z = 0.f;
#pragma unroll
    for (int g = 0; g < 16; ++g) z += red_z[g][tid];
    const size_t o = ((size_t)n * NBLK + blk) * L_ + tid;
    pm[o] = m_l[tid];
    pz[o] = z;
  }

  // ---- Phase F: partial out = e^T (16 x 256) @ b (256 x 128); each wave owns one N-tile ----
  v8f oa = (v8f){};
  const int dbase = wv * 16;
#pragma unroll
  for (int k = 0; k < 8; ++k) {
    const int k0 = k * 32;
    v16bf aa = load_a_bf16(&e_t[r][0], k0, hv);          // A: row l=r, K=pixels
    v16bf bb = load_b_bf16(&b_dp[dbase + r][0], k0, hv); // B: col d, K=pixels
    oa = WMMA_BF16(aa, bb, oa);
  }
  float* dst = pacc + ((size_t)n * NBLK + blk) * (L_ * D_);
#pragma unroll
  for (int j = 0; j < 8; ++j) {
    const int l = j + hv * 8;
    dst[l * D_ + dbase + r] = oa[j];
  }
}

// ---------------- final cross-block softmax combine ----------------
__global__ void __launch_bounds__(256)
reduce_kernel(const float* __restrict__ pm, const float* __restrict__ pz,
              const float* __restrict__ pacc, float* __restrict__ out) {
  const int n   = blockIdx.x;
  const int tid = threadIdx.x;
  __shared__ float Ml[L_], Dn[L_];
  if (tid < L_) {
    float m = -3.4e38f;
    for (int b = 0; b < NBLK; ++b)
      m = fmaxf(m, pm[((size_t)n * NBLK + b) * L_ + tid]);
    float den = 0.f;
    for (int b = 0; b < NBLK; ++b) {
      const size_t o = ((size_t)n * NBLK + b) * L_ + tid;
      den += pz[o] * __expf(pm[o] - m);
    }
    Ml[tid] = m;
    Dn[tid] = den;
  }
  __syncthreads();
  const int l  = tid >> 4;            // 8 outputs per thread, all same l
  const int d0 = (tid * 8) & (D_ - 1);
  const float m = Ml[l], den = Dn[l];
  float s[8] = {0.f, 0.f, 0.f, 0.f, 0.f, 0.f, 0.f, 0.f};
  for (int b = 0; b < NBLK; ++b) {
    const float f = __expf(pm[((size_t)n * NBLK + b) * L_ + l] - m);
    const float* ap = pacc + ((size_t)n * NBLK + b) * (L_ * D_) + l * D_ + d0;
#pragma unroll
    for (int i = 0; i < 8; ++i) s[i] = fmaf(f, ap[i], s[i]);
  }
  float* o = out + (size_t)n * (L_ * D_) + l * D_ + d0;
#pragma unroll
  for (int i = 0; i < 8; ++i) o[i] = s[i] / den;
}

// ---------------- host launcher ----------------
extern "C" void kernel_launch(void* const* d_in, const int* in_sizes, int n_in,
                              void* d_out, int out_size, void* d_ws, size_t ws_size,
                              hipStream_t stream) {
  const float* x  = (const float*)d_in[0];
  const float* t  = (const float*)d_in[1];
  const float* W1 = (const float*)d_in[2];
  const float* b1 = (const float*)d_in[3];
  const float* W2 = (const float*)d_in[4];
  const float* b2 = (const float*)d_in[5];

  char* ws = (char*)d_ws;
  __bf16* w2bf = (__bf16*)(ws + 0);          //  65536 B : W2 in bf16, [d][c]
  __bf16* a_bf = (__bf16*)(ws + 65536);      // 131072 B : a in bf16, [n][l][e]
  float*  pm   = (float*)(ws + 196608);      // 131072 B : per-block max  [n*NBLK][L]
  float*  pz   = (float*)(ws + 327680);      // 131072 B : per-block sum  [n*NBLK][L]
  float*  pacc = (float*)(ws + 458752);      // 16.8 MB  : per-block e^T b [n*NBLK][L][D]

  conv_w2_kernel<<<dim3((D_ * C_) / 256), dim3(256), 0, stream>>>(W2, w2bf);
  prep_a_kernel<<<dim3((N_ * L_ * D_) / 256), dim3(256), 0, stream>>>(t, W1, b1, a_bf);

  main_kernel<<<dim3(NBLK, N_), dim3(256), 0, stream>>>(x, b2, w2bf, a_bf, pm, pz, pacc);
  reduce_kernel<<<dim3(N_), dim3(256), 0, stream>>>(pm, pz, pacc, (float*)d_out);
}
